// LocalAttention_49941879718190
// MI455X (gfx1250) — matrix operand order repaired and checked by reading, not compiled, not run on hardware
//
#include <hip/hip_runtime.h>

// ---------------- types ----------------
typedef __bf16 bf16_t;
typedef __attribute__((ext_vector_type(4)))  bf16_t v4bf;
typedef __attribute__((ext_vector_type(8)))  bf16_t v8bf;
typedef __attribute__((ext_vector_type(16))) bf16_t v16bf;
typedef __attribute__((ext_vector_type(8)))  float  v8f;

union V16U { v16bf v; v8bf h[2]; };
static __device__ __forceinline__ v16bf mk16(v8bf lo, v8bf hi) {
    V16U u; u.h[0] = lo; u.h[1] = hi; return u.v;
}
static __device__ __forceinline__ v8f wmma_bf16(v16bf a, v16bf b, v8f c) {
    return __builtin_amdgcn_wmma_f32_16x16x32_bf16(false, a, false, b, (short)0, c, false, false);
}

// problem constants
#define BATCH 2
#define SEQ   2048
#define DIM   1024
#define HEADS 16
#define DH    64
#define HALFW 64          // window_size/2
#define M_TOT (BATCH*SEQ) // 4096
#define N_QKV (3*DIM)     // 3072

// ---------------- kernel 1: RMSNorm -> bf16 ----------------
__global__ void __launch_bounds__(256)
k_rmsnorm(const float* __restrict__ x, const float* __restrict__ wn, bf16_t* __restrict__ xn) {
    int row = blockIdx.x;           // 0..4095
    int tid = threadIdx.x;          // 256 threads, 4 f32 each
    const float4* xr = (const float4*)(x + (size_t)row * DIM);
    float4 v = xr[tid];
    float ss = v.x*v.x + v.y*v.y + v.z*v.z + v.w*v.w;
    __shared__ float red[256];
    red[tid] = ss; __syncthreads();
    for (int s = 128; s > 0; s >>= 1) { if (tid < s) red[tid] += red[tid + s]; __syncthreads(); }
    float inv = rsqrtf(red[0] * (1.0f / (float)DIM) + 1e-6f);
    float4 w = ((const float4*)wn)[tid];
    v4bf o = { (bf16_t)(v.x*inv*w.x), (bf16_t)(v.y*inv*w.y),
               (bf16_t)(v.z*inv*w.z), (bf16_t)(v.w*inv*w.w) };
    *(v4bf*)(xn + (size_t)row * DIM + tid * 4) = o;
}

// ---------------- kernel 2: f32 -> bf16 convert ----------------
__global__ void __launch_bounds__(256)
k_cvt(const float* __restrict__ src, bf16_t* __restrict__ dst, int n4) {
    int i = blockIdx.x * 256 + threadIdx.x;
    if (i < n4) {
        float4 v = ((const float4*)src)[i];
        v4bf o = { (bf16_t)v.x, (bf16_t)v.y, (bf16_t)v.z, (bf16_t)v.w };
        ((v4bf*)dst)[i] = o;
    }
}

// ---------------- kernel 3: QKV GEMM + scatter ----------------
// y[m,n] = sum_k xn[m,k] * Wqkv[n,k];  64x32 tile per wave
__global__ void __launch_bounds__(256)
k_qkv(const bf16_t* __restrict__ xn, const bf16_t* __restrict__ w,
      bf16_t* __restrict__ qb, bf16_t* __restrict__ kb, bf16_t* __restrict__ vt) {
    int lane = threadIdx.x & 31;
    int wid  = __builtin_amdgcn_readfirstlane((blockIdx.x * 256 + (int)threadIdx.x) >> 5);
    int mt = wid / (N_QKV / 32), nt = wid % (N_QKV / 32);
    int m0 = mt * 64, n0 = nt * 32;
    int half = lane >> 4, ln = lane & 15;

    const bf16_t* arow[4]; const bf16_t* brow[2];
#pragma unroll
    for (int i = 0; i < 4; ++i) arow[i] = xn + (size_t)(m0 + 16*i + ln) * DIM;
#pragma unroll
    for (int j = 0; j < 2; ++j) brow[j] = w + (size_t)(n0 + 16*j + ln) * DIM;

    v8f acc[4][2] = {};
    for (int k0 = 0; k0 < DIM; k0 += 32) {
        __builtin_prefetch(arow[0] + k0 + 128, 0, 1);
        v16bf a[4], b[2];
#pragma unroll
        for (int i = 0; i < 4; ++i)
            a[i] = mk16(*(const v8bf*)(arow[i] + k0 + half*16),
                        *(const v8bf*)(arow[i] + k0 + half*16 + 8));
#pragma unroll
        for (int j = 0; j < 2; ++j)
            b[j] = mk16(*(const v8bf*)(brow[j] + k0 + half*16),
                        *(const v8bf*)(brow[j] + k0 + half*16 + 8));
#pragma unroll
        for (int i = 0; i < 4; ++i)
#pragma unroll
            for (int j = 0; j < 2; ++j)
                acc[i][j] = wmma_bf16(a[i], b[j], acc[i][j]);
    }
#pragma unroll
    for (int i = 0; i < 4; ++i)
#pragma unroll
    for (int j = 0; j < 2; ++j)
#pragma unroll
    for (int r = 0; r < 8; ++r) {
        int m = m0 + 16*i + r + 8*half;
        int n = n0 + 16*j + ln;
        bf16_t bv = (bf16_t)acc[i][j][r];
        int which = n >> 10, d = n & 1023;
        int h = d >> 6, c = d & 63;
        int bb = m >> 11, l = m & (SEQ - 1);
        size_t hoff = ((size_t)bb * HEADS + h);
        if (which == 0)      qb[(hoff * SEQ + l) * DH + c] = bv;
        else if (which == 1) kb[(hoff * SEQ + l) * DH + c] = bv;
        else                 vt[(hoff * DH + c) * SEQ + l] = bv;   // V transposed
    }
}

// ---------------- kernel 4: banded attention ----------------
// one wave per (b, h, 16-query tile); 4 waves per block
#define NW 4
#define NT 9            // key tiles in band
#define KPAD 160        // 9*16 padded to multiple of 32
__global__ void __launch_bounds__(128)
k_attn(const bf16_t* __restrict__ qb, const bf16_t* __restrict__ kb,
       const bf16_t* __restrict__ vt, bf16_t* __restrict__ ao) {
    __shared__ float sc[NW][16][148];                    // score rows (padded pitch)
    __shared__ __align__(16) bf16_t pb[NW][16][KPAD];    // probs as A-fragment source
    int tid = threadIdx.x, lane = tid & 31, w = tid >> 5;
    int wid = __builtin_amdgcn_readfirstlane(blockIdx.x * NW + w);
    int qt = wid & 127, h = (wid >> 7) & 15, b = wid >> 11;
    int q0 = qt * 16;
    int half = lane >> 4, ln = lane & 15;
    size_t headoff = ((size_t)b * HEADS + h) * (size_t)SEQ * DH;

    // Q fragments (A: 16 x 64, two K=32 slabs), loaded once
    const bf16_t* qrow = qb + headoff + (size_t)(q0 + ln) * DH;
    v16bf qf[2];
#pragma unroll
    for (int f = 0; f < 2; ++f)
        qf[f] = mk16(*(const v8bf*)(qrow + 32*f + half*16),
                     *(const v8bf*)(qrow + 32*f + half*16 + 8));

    int jbase = q0 - HALFW;
    for (int t = 0; t < NT; ++t) {
        int jstart = jbase + 16 * t;                 // wave-uniform
        if (jstart >= 0 && jstart < SEQ) {
            const bf16_t* krow = kb + headoff + (size_t)(jstart + ln) * DH;
            v8f s = {};
#pragma unroll
            for (int f = 0; f < 2; ++f) {
                v16bf kf = mk16(*(const v8bf*)(krow + 32*f + half*16),
                                *(const v8bf*)(krow + 32*f + half*16 + 8));
                s = wmma_bf16(qf[f], kf, s);
            }
#pragma unroll
            for (int r = 0; r < 8; ++r) {
                int i = q0 + r + 8*half, j = jstart + ln;
                float v = s[r] * 0.125f;             // 1/sqrt(64)
                int dd = i - j;
                if (dd < -HALFW || dd > HALFW) v = -1e30f;
                sc[w][r + 8*half][16*t + ln] = v;
            }
        } else {
#pragma unroll
            for (int r = 0; r < 8; ++r) sc[w][r + 8*half][16*t + ln] = -1e30f;
        }
    }
    __syncthreads();
    if (lane < 16) {                                 // one row per lane
        float* row = sc[w][lane];
        float m = -1e30f;
        for (int c = 0; c < NT*16; ++c) m = fmaxf(m, row[c]);
        float sum = 0.f;
        for (int c = 0; c < NT*16; ++c) { float e = __expf(row[c] - m); row[c] = e; sum += e; }
        float inv = 1.0f / sum;
        bf16_t* prow = pb[w][lane];
        for (int c = 0; c < NT*16; ++c) prow[c] = (bf16_t)(row[c] * inv);
        for (int c = NT*16; c < KPAD; ++c) prow[c] = (bf16_t)0.0f;
    }
    __syncthreads();

    // out tile 16 x 64 = P(16xKPAD) @ Vt(KPAD x 64)
    const bf16_t* prow = pb[w][ln];
#pragma unroll
    for (int nt2 = 0; nt2 < 4; ++nt2) {
        int c0 = nt2 * 16;
        const bf16_t* vcol = vt + headoff + (size_t)(c0 + ln) * SEQ;
        v8f o = {};
#pragma unroll
        for (int kt = 0; kt < KPAD/32; ++kt) {
            v16bf pf = mk16(*(const v8bf*)(prow + 32*kt + half*16),
                            *(const v8bf*)(prow + 32*kt + half*16 + 8));
            int j0 = jbase + 32*kt + 16*half;        // 16-aligned chunk
            if (j0 < 0 || j0 > SEQ - 16) j0 = 0;     // garbage * P==0
            v16bf vf = mk16(*(const v8bf*)(vcol + j0),
                            *(const v8bf*)(vcol + j0 + 8));
            o = wmma_bf16(pf, vf, o);
        }
#pragma unroll
        for (int r = 0; r < 8; ++r)
            ao[((size_t)b * SEQ + q0 + r + 8*half) * DIM + h*DH + c0 + ln] = (bf16_t)o[r];
    }
}

// ---------------- kernel 5: output projection + skip ----------------
__global__ void __launch_bounds__(256)
k_out(const bf16_t* __restrict__ ao, const bf16_t* __restrict__ w,
      const float* __restrict__ x, float* __restrict__ out) {
    int lane = threadIdx.x & 31;
    int wid  = __builtin_amdgcn_readfirstlane((blockIdx.x * 256 + (int)threadIdx.x) >> 5);
    int mt = wid / (DIM / 32), nt = wid % (DIM / 32);
    int m0 = mt * 64, n0 = nt * 32;
    int half = lane >> 4, ln = lane & 15;

    const bf16_t* arow[4]; const bf16_t* brow[2];
#pragma unroll
    for (int i = 0; i < 4; ++i) arow[i] = ao + (size_t)(m0 + 16*i + ln) * DIM;
#pragma unroll
    for (int j = 0; j < 2; ++j) brow[j] = w + (size_t)(n0 + 16*j + ln) * DIM;

    v8f acc[4][2] = {};
    for (int k0 = 0; k0 < DIM; k0 += 32) {
        __builtin_prefetch(arow[0] + k0 + 128, 0, 1);
        v16bf a[4], b[2];
#pragma unroll
        for (int i = 0; i < 4; ++i)
            a[i] = mk16(*(const v8bf*)(arow[i] + k0 + half*16),
                        *(const v8bf*)(arow[i] + k0 + half*16 + 8));
#pragma unroll
        for (int j = 0; j < 2; ++j)
            b[j] = mk16(*(const v8bf*)(brow[j] + k0 + half*16),
                        *(const v8bf*)(brow[j] + k0 + half*16 + 8));
#pragma unroll
        for (int i = 0; i < 4; ++i)
#pragma unroll
            for (int j = 0; j < 2; ++j)
                acc[i][j] = wmma_bf16(a[i], b[j], acc[i][j]);
    }
#pragma unroll
    for (int i = 0; i < 4; ++i)
#pragma unroll
    for (int j = 0; j < 2; ++j)
#pragma unroll
    for (int r = 0; r < 8; ++r) {
        int m = m0 + 16*i + r + 8*half;
        int n = n0 + 16*j + ln;
        size_t idx = (size_t)m * DIM + n;
        out[idx] = acc[i][j][r] + x[idx];            // fused skip
    }
}

// ---------------- launcher ----------------
extern "C" void kernel_launch(void* const* d_in, const int* in_sizes, int n_in,
                              void* d_out, int out_size, void* d_ws, size_t ws_size,
                              hipStream_t stream) {
    const float* x    = (const float*)d_in[0];
    const float* wn   = (const float*)d_in[1];
    const float* wqkv = (const float*)d_in[2];
    const float* wout = (const float*)d_in[3];
    float* out = (float*)d_out;

    char* ws = (char*)d_ws;
    const size_t MB = 1u << 20;
    bf16_t* xn    = (bf16_t*)(ws + 0*MB);   // 8 MB
    bf16_t* wqkvb = (bf16_t*)(ws + 8*MB);   // 6 MB
    bf16_t* woutb = (bf16_t*)(ws + 14*MB);  // 2 MB
    bf16_t* qb    = (bf16_t*)(ws + 16*MB);  // 8 MB  [B,H,L,dh]
    bf16_t* kb    = (bf16_t*)(ws + 24*MB);  // 8 MB  [B,H,L,dh]
    bf16_t* vt    = (bf16_t*)(ws + 32*MB);  // 8 MB  [B,H,dh,L]
    bf16_t* ao    = (bf16_t*)(ws + 40*MB);  // 8 MB  [B,L,D]

    k_rmsnorm<<<M_TOT, 256, 0, stream>>>(x, wn, xn);
    k_cvt<<<(N_QKV*DIM/4 + 255)/256, 256, 0, stream>>>(wqkv, wqkvb, N_QKV*DIM/4);
    k_cvt<<<(DIM*DIM/4 + 255)/256, 256, 0, stream>>>(wout, woutb, DIM*DIM/4);
    k_qkv<<<(M_TOT/64)*(N_QKV/32)/8, 256, 0, stream>>>(xn, wqkvb, qb, kb, vt);
    k_attn<<<(BATCH*HEADS*(SEQ/16))/NW, 128, 0, stream>>>(qb, kb, vt, ao);
    k_out<<<(M_TOT/64)*(DIM/32)/8, 256, 0, stream>>>(ao, woutb, x, out);
}
